// DeepseekV2AttentionMLA_65515431133682
// MI455X (gfx1250) — matrix operand order repaired and checked by reading, not compiled, not run on hardware
//
#include <hip/hip_runtime.h>
#include <hip/hip_bf16.h>
#include <cstdint>

typedef __bf16 bf16;
typedef __bf16 v16bf __attribute__((ext_vector_type(16)));
typedef float  v8f   __attribute__((ext_vector_type(8)));

union Frag16 { v16bf v; uint4 q[2]; };

// ---------------- constants -------------------------------------------------
static constexpr int SEQ  = 1024;
static constexpr int HIDC = 5120;
static constexpr int NHC  = 128;
static constexpr int DNC  = 128;
static constexpr int DRC  = 64;
static constexpr int DVC  = 128;
static constexpr int QLRC = 1536;
static constexpr int KVLRC= 512;
static constexpr int QKV_N = QLRC + KVLRC + DRC;        // 2112
static constexpr int QB_N  = NHC * (DNC + DRC);         // 24576
static constexpr int KVB_N = NHC * (DNC + DVC);         // 32768
static constexpr int O_N   = NHC * DVC;                 // 16384
static constexpr int DQK   = DNC + DRC;                 // 192

// ---------------- fp32 -> bf16 convert --------------------------------------
__global__ __launch_bounds__(256) void f32_to_bf16_kernel(
    const float* __restrict__ x, bf16* __restrict__ y, int n)
{
    for (int i = blockIdx.x * blockDim.x + threadIdx.x; i < n;
         i += gridDim.x * blockDim.x)
        y[i] = (bf16)x[i];
}

// ---------------- RMSNorm (fp32 in, bf16 out, weight fused) -----------------
__global__ __launch_bounds__(256) void rmsnorm_bf16_kernel(
    const float* __restrict__ x, int ldx, int W,
    const float* __restrict__ w, bf16* __restrict__ y, int ldy)
{
    const int t = blockIdx.x;
    const float* xr = x + (size_t)t * ldx;
    float ss = 0.0f;
    for (int i = threadIdx.x; i < W; i += 256) { float v = xr[i]; ss += v * v; }
    __shared__ float red[256];
    red[threadIdx.x] = ss;
    __syncthreads();
    for (int s = 128; s > 0; s >>= 1) {
        if (threadIdx.x < s) red[threadIdx.x] += red[threadIdx.x + s];
        __syncthreads();
    }
    const float rstd = rsqrtf(red[0] / (float)W + 1e-6f);
    for (int i = threadIdx.x; i < W; i += 256)
        y[(size_t)t * ldy + i] = (bf16)(xr[i] * rstd * w[i]);
}

// ---------------- generic bf16 WMMA GEMM ------------------------------------
// C[M,N] = A[M,K](bf16,row-major) * B[K,N](fp32,row-major, converted to bf16)
// block = 256 threads = 8 waves (wave32), wave grid 2(M) x 4(N), BK = 32.
// BM=256 halves HBM weight re-traffic vs BM=128 (B is streamed M/BM times).
template<int BM, int BN, bool OUT_BF16>
__global__ __launch_bounds__(256) void gemm_bf16_kernel(
    const bf16* __restrict__ A, const float* __restrict__ B,
    void* __restrict__ Cvoid, int M, int N, int K)
{
    constexpr int BK = 32;
    constexpr int WM = BM / 2;
    constexpr int WN = BN / 4;
    constexpr int MT = WM / 16;
    constexpr int NT = WN / 16;

    __shared__ bf16 As[BM * BK];   // [m][k]
    __shared__ bf16 Bs[BN * BK];   // [n][k] (transposed for contiguous B frags)

    const int tid   = threadIdx.x;
    const int wave  = tid >> 5;
    const int lane  = tid & 31;
    const int lmod  = lane & 15;
    const int lhalf = lane >> 4;
    const int wm = wave >> 2;
    const int wn = wave & 3;
    const int bm = blockIdx.y * BM;
    const int bn = blockIdx.x * BN;

    v8f acc[MT][NT];
    #pragma unroll
    for (int i = 0; i < MT; ++i)
        #pragma unroll
        for (int j = 0; j < NT; ++j)
            #pragma unroll
            for (int r = 0; r < 8; ++r) acc[i][j][r] = 0.0f;

    for (int k0 = 0; k0 < K; k0 += BK) {
        // A tile: BM x 32 bf16, copied in uint4 (8 bf16) units
        for (int idx = tid; idx < BM * 4; idx += 256) {
            const int row = idx >> 2, qd = idx & 3;
            *reinterpret_cast<uint4*>(&As[row * BK + qd * 8]) =
                *reinterpret_cast<const uint4*>(A + (size_t)(bm + row) * K + k0 + qd * 8);
        }
        // B tile: 32 x BN fp32 -> bf16, stored transposed [n][k]
        for (int idx = tid; idx < (BK * BN) / 4; idx += 256) {
            const int k  = idx / (BN / 4);
            const int c4 = idx % (BN / 4);
            const float4 f = *reinterpret_cast<const float4*>(
                B + (size_t)(k0 + k) * N + bn + c4 * 4);
            const int c = c4 * 4;
            Bs[(c + 0) * BK + k] = (bf16)f.x;
            Bs[(c + 1) * BK + k] = (bf16)f.y;
            Bs[(c + 2) * BK + k] = (bf16)f.z;
            Bs[(c + 3) * BK + k] = (bf16)f.w;
        }
        __syncthreads();

        // B fragments held across the i-loop; exactly one A fragment live at
        // a time -> MT=8 fits without blowing the VGPR budget.
        Frag16 bfr[NT];
        #pragma unroll
        for (int j = 0; j < NT; ++j) {
            const int col = wn * WN + j * 16 + lmod;
            bfr[j].q[0] = *reinterpret_cast<const uint4*>(&Bs[col * BK + lhalf * 8]);
            bfr[j].q[1] = *reinterpret_cast<const uint4*>(&Bs[col * BK + 16 + lhalf * 8]);
        }
        #pragma unroll
        for (int i = 0; i < MT; ++i) {
            Frag16 af;
            const int row = wm * WM + i * 16 + lmod;
            af.q[0] = *reinterpret_cast<const uint4*>(&As[row * BK + lhalf * 8]);
            af.q[1] = *reinterpret_cast<const uint4*>(&As[row * BK + 16 + lhalf * 8]);
            #pragma unroll
            for (int j = 0; j < NT; ++j)
                acc[i][j] = __builtin_amdgcn_wmma_f32_16x16x32_bf16(
                    false, af.v, false, bfr[j].v, (short)0, acc[i][j], false, false);
        }
        __syncthreads();
    }

    #pragma unroll
    for (int i = 0; i < MT; ++i)
        #pragma unroll
        for (int j = 0; j < NT; ++j)
            #pragma unroll
            for (int r = 0; r < 8; ++r) {
                const int row = bm + wm * WM + i * 16 + lhalf * 8 + r;
                const int col = bn + wn * WN + j * 16 + lmod;
                if (OUT_BF16)
                    reinterpret_cast<bf16*>(Cvoid)[(size_t)row * N + col] = (bf16)acc[i][j][r];
                else
                    reinterpret_cast<float*>(Cvoid)[(size_t)row * N + col] = acc[i][j][r];
            }
}

// ---------------- RoPE: rotate q_pe in place (bf16) + emit roped k_pe -------
__global__ __launch_bounds__(256) void rope_kernel(
    const int* __restrict__ pos, const float* __restrict__ qkv,
    bf16* __restrict__ qbf, bf16* __restrict__ kpe)
{
    const int t = blockIdx.x;
    __shared__ float cs[32], sn[32];
    if (threadIdx.x < 32) {
        const int p = threadIdx.x;
        const float invf = __powf(10000.0f, -(float)(2 * p) / 64.0f);
        const float fr = (float)pos[t] * invf;
        cs[p] = __cosf(fr);
        sn[p] = __sinf(fr);
    }
    __syncthreads();
    if (threadIdx.x < 32) {
        const int p = threadIdx.x;
        const float x1 = qkv[(size_t)t * QKV_N + QLRC + KVLRC + 2 * p];
        const float x2 = qkv[(size_t)t * QKV_N + QLRC + KVLRC + 2 * p + 1];
        kpe[(size_t)t * DRC + 2 * p]     = (bf16)(x1 * cs[p] - x2 * sn[p]);
        kpe[(size_t)t * DRC + 2 * p + 1] = (bf16)(x2 * cs[p] + x1 * sn[p]);
    }
    for (int idx = threadIdx.x; idx < NHC * 32; idx += 256) {
        const int h = idx >> 5, p = idx & 31;
        const size_t base = (size_t)t * QB_N + (size_t)h * DQK + DNC + 2 * p;
        const float x1 = (float)qbf[base];
        const float x2 = (float)qbf[base + 1];
        qbf[base]     = (bf16)(x1 * cs[p] - x2 * sn[p]);
        qbf[base + 1] = (bf16)(x2 * cs[p] + x1 * sn[p]);
    }
}

// ---------------- Flash attention, one block per (head, 128-row q block) ----
__global__ __launch_bounds__(256) void mla_attn_kernel(
    const bf16* __restrict__ qbf,   // [SEQ][NH*192]
    const bf16* __restrict__ kvbf,  // [SEQ][NH*256]  (k_nope | v)
    const bf16* __restrict__ kpe,   // [SEQ][64]
    bf16* __restrict__ obf)         // [SEQ][NH*128]
{
    constexpr int BKV = 32;
    constexpr float SCALE = 0.07216878364870322f; // 192^-0.5

    const int h  = blockIdx.x;
    const int qb = blockIdx.y;
    const int tid   = threadIdx.x;
    const int wave  = tid >> 5;
    const int lane  = tid & 31;
    const int lmod  = lane & 15;
    const int lhalf = lane >> 4;

    __shared__ bf16 Ks[BKV * DQK];       // [key][d]       12 KB
    __shared__ bf16 Vt[DVC * BKV];       // [d][key]        8 KB
    __shared__ bf16 Plds[8][16 * BKV];   // per-wave P      8 KB

    const int q0 = qb * 128 + wave * 16;

    // Q fragments: 6 x (16x32) bf16, A-matrix layout, straight from global
    Frag16 qf[6];
    const bf16* qrow = qbf + (size_t)(q0 + lmod) * QB_N + (size_t)h * DQK;
    #pragma unroll
    for (int f = 0; f < 6; ++f) {
        const int d0 = f * 32;
        qf[f].q[0] = *reinterpret_cast<const uint4*>(qrow + d0 + lhalf * 8);
        qf[f].q[1] = *reinterpret_cast<const uint4*>(qrow + d0 + 16 + lhalf * 8);
    }

    v8f oacc[8];
    #pragma unroll
    for (int t = 0; t < 8; ++t)
        #pragma unroll
        for (int r = 0; r < 8; ++r) oacc[t][r] = 0.0f;
    float mrow[8], lrow[8];
    #pragma unroll
    for (int r = 0; r < 8; ++r) { mrow[r] = -1e30f; lrow[r] = 0.0f; }

    const int kend = (qb + 1) * 128;
    for (int k0 = 0; k0 < kend; k0 += BKV) {
        // K tile [32][192]: k_nope from kvbf, k_pe shared across heads
        for (int idx = tid; idx < 32 * 12; idx += 256) {
            const int key = idx / 12, c = idx % 12, d0 = c * 16;
            const bf16* src = (d0 < DNC)
                ? kvbf + (size_t)(k0 + key) * KVB_N + (size_t)h * (DNC + DVC) + d0
                : kpe + (size_t)(k0 + key) * DRC + (d0 - DNC);
            *reinterpret_cast<uint4*>(&Ks[key * DQK + d0])     = reinterpret_cast<const uint4*>(src)[0];
            *reinterpret_cast<uint4*>(&Ks[key * DQK + d0 + 8]) = reinterpret_cast<const uint4*>(src)[1];
        }
        // V tile, transposed to [d][key]
        for (int idx = tid; idx < 32 * 8; idx += 256) {
            const int key = idx / 8, c = idx % 8, d0 = c * 16;
            const bf16* src = kvbf + (size_t)(k0 + key) * KVB_N + (size_t)h * (DNC + DVC) + DNC + d0;
            #pragma unroll
            for (int e = 0; e < 16; ++e) Vt[(d0 + e) * BKV + key] = src[e];
        }
        __syncthreads();

        // S = Q K^T  (16 x 32 scores per wave)
        v8f sacc[2];
        #pragma unroll
        for (int jt = 0; jt < 2; ++jt)
            #pragma unroll
            for (int r = 0; r < 8; ++r) sacc[jt][r] = 0.0f;
        #pragma unroll
        for (int jt = 0; jt < 2; ++jt)
            #pragma unroll
            for (int f = 0; f < 6; ++f) {
                Frag16 kf;
                const bf16* kb = &Ks[(jt * 16 + lmod) * DQK + f * 32 + lhalf * 8];
                kf.q[0] = *reinterpret_cast<const uint4*>(kb);
                kf.q[1] = *reinterpret_cast<const uint4*>(kb + 16);
                sacc[jt] = __builtin_amdgcn_wmma_f32_16x16x32_bf16(
                    false, qf[f].v, false, kf.v, (short)0, sacc[jt], false, false);
            }

        // scale + causal mask + online softmax stats
        float mnew[8];
        #pragma unroll
        for (int r = 0; r < 8; ++r) mnew[r] = mrow[r];
        #pragma unroll
        for (int jt = 0; jt < 2; ++jt)
            #pragma unroll
            for (int r = 0; r < 8; ++r) {
                float s = sacc[jt][r] * SCALE;
                const int ki = k0 + jt * 16 + lmod;
                const int qi = q0 + lhalf * 8 + r;
                s = (ki <= qi) ? s : -1e30f;
                sacc[jt][r] = s;
                mnew[r] = fmaxf(mnew[r], s);
            }
        #pragma unroll
        for (int m = 1; m < 16; m <<= 1)
            #pragma unroll
            for (int r = 0; r < 8; ++r)
                mnew[r] = fmaxf(mnew[r], __shfl_xor(mnew[r], m, 32));

        float resc[8], psum[8];
        #pragma unroll
        for (int r = 0; r < 8; ++r) {
            resc[r] = __expf(mrow[r] - mnew[r]);
            psum[r] = 0.0f;
        }
        #pragma unroll
        for (int jt = 0; jt < 2; ++jt)
            #pragma unroll
            for (int r = 0; r < 8; ++r) {
                const float p = __expf(sacc[jt][r] - mnew[r]);
                psum[r] += p;
                Plds[wave][(lhalf * 8 + r) * BKV + jt * 16 + lmod] = (bf16)p;
            }
        #pragma unroll
        for (int m = 1; m < 16; m <<= 1)
            #pragma unroll
            for (int r = 0; r < 8; ++r)
                psum[r] += __shfl_xor(psum[r], m, 32);
        #pragma unroll
        for (int r = 0; r < 8; ++r) {
            lrow[r] = lrow[r] * resc[r] + psum[r];
            mrow[r] = mnew[r];
        }
        #pragma unroll
        for (int t = 0; t < 8; ++t)
            #pragma unroll
            for (int r = 0; r < 8; ++r) oacc[t][r] *= resc[r];

        // O += P V   (P re-layouted C->A via per-wave LDS)
        Frag16 pf;
        pf.q[0] = *reinterpret_cast<const uint4*>(&Plds[wave][lmod * BKV + lhalf * 8]);
        pf.q[1] = *reinterpret_cast<const uint4*>(&Plds[wave][lmod * BKV + 16 + lhalf * 8]);
        #pragma unroll
        for (int t = 0; t < 8; ++t) {
            Frag16 vf;
            const bf16* vb = &Vt[(t * 16 + lmod) * BKV + lhalf * 8];
            vf.q[0] = *reinterpret_cast<const uint4*>(vb);
            vf.q[1] = *reinterpret_cast<const uint4*>(vb + 16);
            oacc[t] = __builtin_amdgcn_wmma_f32_16x16x32_bf16(
                false, pf.v, false, vf.v, (short)0, oacc[t], false, false);
        }
        __syncthreads();
    }

    // finalize
    #pragma unroll
    for (int t = 0; t < 8; ++t)
        #pragma unroll
        for (int r = 0; r < 8; ++r) {
            const int row = q0 + lhalf * 8 + r;
            const int col = t * 16 + lmod;
            obf[(size_t)row * O_N + (size_t)h * DVC + col] = (bf16)(oacc[t][r] / lrow[r]);
        }
}

// ---------------- launcher --------------------------------------------------
extern "C" void kernel_launch(void* const* d_in, const int* in_sizes, int n_in,
                              void* d_out, int out_size, void* d_ws, size_t ws_size,
                              hipStream_t stream) {
    const int*   positions = (const int*)  d_in[0];
    const float* hidden    = (const float*)d_in[1];
    const float* w_qkv_a   = (const float*)d_in[2];
    const float* q_a_ln_w  = (const float*)d_in[3];
    const float* w_q_b     = (const float*)d_in[4];
    const float* kv_a_ln_w = (const float*)d_in[5];
    const float* w_kv_b    = (const float*)d_in[6];
    const float* w_o       = (const float*)d_in[7];
    float* out = (float*)d_out;

    auto alignup = [](size_t x) { return (x + 255) & ~(size_t)255; };
    char* base = (char*)d_ws;
    size_t off = 0;
    bf16*  Xh   = (bf16*) (base + off); off += alignup((size_t)SEQ * HIDC  * 2);
    float* qkv  = (float*)(base + off); off += alignup((size_t)SEQ * QKV_N * 4);
    bf16*  qan  = (bf16*) (base + off); off += alignup((size_t)SEQ * QLRC  * 2);
    bf16*  kvan = (bf16*) (base + off); off += alignup((size_t)SEQ * KVLRC * 2);
    bf16*  qbf  = (bf16*) (base + off); off += alignup((size_t)SEQ * QB_N  * 2);
    bf16*  kvbf = (bf16*) (base + off); off += alignup((size_t)SEQ * KVB_N * 2);
    bf16*  kpe  = (bf16*) (base + off); off += alignup((size_t)SEQ * DRC   * 2);
    bf16*  obf  = (bf16*) (base + off); off += alignup((size_t)SEQ * O_N   * 2);
    (void)off; (void)ws_size; (void)in_sizes; (void)n_in; (void)out_size;

    // 1) hidden -> bf16
    f32_to_bf16_kernel<<<2048, 256, 0, stream>>>(hidden, Xh, SEQ * HIDC);

    // 2) qkv = Xh @ w_qkv_a   (1024 x 2112, K=5120), fp32 out
    gemm_bf16_kernel<256, 64, false>
        <<<dim3(QKV_N / 64, SEQ / 256), 256, 0, stream>>>(
            Xh, w_qkv_a, qkv, SEQ, QKV_N, HIDC);

    // 3) RMSNorms -> bf16 activations
    rmsnorm_bf16_kernel<<<SEQ, 256, 0, stream>>>(qkv, QKV_N, QLRC, q_a_ln_w, qan, QLRC);
    rmsnorm_bf16_kernel<<<SEQ, 256, 0, stream>>>(qkv + QLRC, QKV_N, KVLRC, kv_a_ln_w, kvan, KVLRC);

    // 4) q = qan @ w_q_b  (1024 x 24576, K=1536), bf16 out
    gemm_bf16_kernel<256, 128, true>
        <<<dim3(QB_N / 128, SEQ / 256), 256, 0, stream>>>(
            qan, w_q_b, qbf, SEQ, QB_N, QLRC);

    // 5) kv = kvan @ w_kv_b  (1024 x 32768, K=512), bf16 out
    gemm_bf16_kernel<256, 128, true>
        <<<dim3(KVB_N / 128, SEQ / 256), 256, 0, stream>>>(
            kvan, w_kv_b, kvbf, SEQ, KVB_N, KVLRC);

    // 6) RoPE on q_pe (in place) + roped k_pe buffer
    rope_kernel<<<SEQ, 256, 0, stream>>>(positions, qkv, qbf, kpe);

    // 7) flash attention per (head, q-block)
    mla_attn_kernel<<<dim3(NHC, SEQ / 128), 256, 0, stream>>>(qbf, kvbf, kpe, obf);

    // 8) out = obf @ w_o  (1024 x 5120, K=16384), fp32 out
    gemm_bf16_kernel<256, 128, false>
        <<<dim3(5120 / 128, SEQ / 256), 256, 0, stream>>>(
            obf, w_o, out, SEQ, 5120, O_N);
}
